// MultiHeadAttention_39298950759031
// MI455X (gfx1250) — matrix operand order repaired and checked by reading, not compiled, not run on hardware
//
#include <hip/hip_runtime.h>
#include <stdint.h>

typedef __attribute__((ext_vector_type(16))) _Float16 v16h;
typedef __attribute__((ext_vector_type(8)))  float    v8f;

#define D_MODEL 768
#define SEQ     2048
#define BATCH   4
#define NHEAD   12
#define HDIM    64
#define MTOT    (BATCH*SEQ)   /* 8192 */

union FragH { v16h v; uint32_t u[8]; };
union H2    { uint32_t u; _Float16 h[2]; };

// 16-bit A-matrix 16x32 fragment k-index for VGPR v, lane-half hh (ISA 7.12.2)
__device__ __forceinline__ int kmapA(int v, int hh) { return ((v >> 2) << 4) + hh * 8 + ((v & 3) << 1); }
// 16-bit B-matrix 32x16 fragment k-index (same k order as kmapA)
__device__ __forceinline__ int kmapB(int v, int hh) { return ((v >> 2) << 4) + hh * 8 + ((v & 3) << 1); }

#define WMMA_F16(A, B, C) \
    __builtin_amdgcn_wmma_f32_16x16x32_f16(false, (A), false, (B), (short)0, (C), false, false)

// ---------------------------------------------------------------------------
// Kernel 1: fused Q/K/V projections.  X[8192x768] (f32) @ W[768x768] (f32) + b
// -> f16 out in [B,S,H*d] layout.  blockIdx.z in {0,1,2} selects Q/K/V.
// Block tile 128x64, 256 threads = 8 waves (4m x 2n), 32x32 tile per wave
// (2 A-frags x 2 B-frags -> 4 WMMA per 32-k step, full fragment reuse).
// ---------------------------------------------------------------------------
__global__ __launch_bounds__(256)
void qkv_proj_kernel(const float* __restrict__ q_in, const float* __restrict__ k_in,
                     const float* __restrict__ v_in,
                     const float* __restrict__ wq, const float* __restrict__ wk,
                     const float* __restrict__ wv,
                     const float* __restrict__ bq, const float* __restrict__ bk,
                     const float* __restrict__ bv,
                     _Float16* __restrict__ Qo, _Float16* __restrict__ Ko,
                     _Float16* __restrict__ Vo)
{
    const int z = blockIdx.z;
    const float* X  = (z == 0) ? q_in : (z == 1) ? k_in : v_in;
    const float* W  = (z == 0) ? wq   : (z == 1) ? wk   : wv;
    const float* Bb = (z == 0) ? bq   : (z == 1) ? bk   : bv;
    _Float16*    O  = (z == 0) ? Qo   : (z == 1) ? Ko   : Vo;

    __shared__ _Float16 As[128 * 34] __attribute__((aligned(16)));  // [m][k]
    __shared__ _Float16 Bs[ 64 * 34] __attribute__((aligned(16)));  // [n][k] (transposed)

    const int m0  = blockIdx.y * 128;
    const int n0  = blockIdx.x * 64;
    const int tid = threadIdx.x;
    const int wid = tid >> 5, lane = tid & 31, lm = lane & 15, hh = lane >> 4;
    const int wm  = wid & 3,  wn   = wid >> 2;

    v8f c00 = {}, c01 = {}, c10 = {}, c11 = {};

    for (int k0 = 0; k0 < D_MODEL; k0 += 32) {
        // Stage A tile 128x32 (f32 -> f16), coalesced along k
        #pragma unroll
        for (int i = 0; i < 16; i++) {
            int t = tid + i * 256;
            int r = t >> 5, cc = t & 31;
            As[r * 34 + cc] = (_Float16)X[(m0 + r) * D_MODEL + k0 + cc];
        }
        // Stage W tile 32x64 transposed: Bs[n][k] = W[k][n], coalesced along n
        #pragma unroll
        for (int i = 0; i < 8; i++) {
            int t = tid + i * 256;
            int kk = t >> 6, nn = t & 63;
            Bs[nn * 34 + kk] = (_Float16)W[(k0 + kk) * D_MODEL + n0 + nn];
        }
        // Prefetch next k-step tiles (global_prefetch_b8)
        if (k0 + 32 < D_MODEL) {
            __builtin_prefetch(&X[(m0 + (tid >> 1)) * D_MODEL + k0 + 32 + (tid & 1) * 16], 0, 3);
            __builtin_prefetch(&W[(k0 + 32 + (tid >> 3)) * D_MODEL + n0 + (tid & 7) * 8], 0, 3);
        }
        __syncthreads();

        FragH a0, a1, b0, b1;
        #pragma unroll
        for (int v = 0; v < 8; v++) {
            int ka = kmapA(v, hh), kb = kmapB(v, hh);
            a0.u[v] = *(const uint32_t*)&As[(wm * 32 +      lm) * 34 + ka];
            a1.u[v] = *(const uint32_t*)&As[(wm * 32 + 16 + lm) * 34 + ka];
            b0.u[v] = *(const uint32_t*)&Bs[(wn * 32 +      lm) * 34 + kb];
            b1.u[v] = *(const uint32_t*)&Bs[(wn * 32 + 16 + lm) * 34 + kb];
        }
        c00 = WMMA_F16(a0.v, b0.v, c00);
        c01 = WMMA_F16(a0.v, b1.v, c01);
        c10 = WMMA_F16(a1.v, b0.v, c10);
        c11 = WMMA_F16(a1.v, b1.v, c11);
        __syncthreads();
    }

    const int   col0  = n0 + wn * 32 + lm;
    const int   col1  = col0 + 16;
    const float bias0 = Bb[col0];
    const float bias1 = Bb[col1];
    #pragma unroll
    for (int i = 0; i < 8; i++) {
        int row0 = m0 + wm * 32 + i + 8 * hh;    // C-layout: row = i + 8*(lane/16)
        int row1 = row0 + 16;
        O[row0 * D_MODEL + col0] = (_Float16)(c00[i] + bias0);
        O[row0 * D_MODEL + col1] = (_Float16)(c01[i] + bias1);
        O[row1 * D_MODEL + col0] = (_Float16)(c10[i] + bias0);
        O[row1 * D_MODEL + col1] = (_Float16)(c11[i] + bias1);
    }
}

// ---------------------------------------------------------------------------
// Kernel 2: flash attention per (b, h, 64-query tile).  128 threads = 4 waves,
// each wave owns 16 query rows.  Streams key/value blocks of 64:
// 8 score WMMAs + 8 P.V WMMAs per barrier round per wave.
// ---------------------------------------------------------------------------
__global__ __launch_bounds__(128)
void flash_attn_kernel(const _Float16* __restrict__ Q, const _Float16* __restrict__ K,
                       const _Float16* __restrict__ V, _Float16* __restrict__ C)
{
    __shared__ _Float16 Qs [64 * 66]     __attribute__((aligned(16))); // [q][d]
    __shared__ _Float16 Ks [64 * 66]     __attribute__((aligned(16))); // [key][d]
    __shared__ _Float16 Vts[64 * 66]     __attribute__((aligned(16))); // [d][key]
    __shared__ _Float16 Ps [4 * 16 * 66] __attribute__((aligned(16))); // per-wave P 16x64

    const int b   = blockIdx.y / NHEAD;
    const int h   = blockIdx.y % NHEAD;
    const int q0  = blockIdx.x * 64;
    const int tid = threadIdx.x;
    const int wid = tid >> 5, lane = tid & 31, lm = lane & 15, hh = lane >> 4;

    // Stage Q tile [64 x 64] f16, half2 granularity
    #pragma unroll
    for (int i = 0; i < 16; i++) {
        int t = tid + i * 128;
        int r = t >> 5, u = t & 31;
        *(uint32_t*)&Qs[r * 66 + u * 2] =
            *(const uint32_t*)&Q[(b * SEQ + q0 + r) * D_MODEL + h * HDIM + u * 2];
    }
    __syncthreads();

    // Q A-fragments (two k-steps over d=64), held in registers for whole loop
    FragH qa[2];
    #pragma unroll
    for (int kk = 0; kk < 2; kk++)
        #pragma unroll
        for (int v = 0; v < 8; v++)
            qa[kk].u[v] = *(const uint32_t*)&Qs[(wid * 16 + lm) * 66 + kk * 32 + kmapA(v, hh)];

    v8f   acc[4] = {};
    float mrow[8], lrow[8];
    #pragma unroll
    for (int i = 0; i < 8; i++) { mrow[i] = -1e30f; lrow[i] = 0.0f; }

    for (int kb = 0; kb < SEQ; kb += 64) {
        __syncthreads();   // previous iteration's LDS reads complete
        // Stage K block [64 x 64] and V block transposed [64 x 64]
        #pragma unroll
        for (int i = 0; i < 16; i++) {
            int t = tid + i * 128;
            int r = t >> 5, u = t & 31;
            int gbase = (b * SEQ + kb + r) * D_MODEL + h * HDIM + u * 2;
            *(uint32_t*)&Ks[r * 66 + u * 2] = *(const uint32_t*)&K[gbase];
            H2 w; w.u = *(const uint32_t*)&V[gbase];
            Vts[(u * 2 + 0) * 66 + r] = w.h[0];
            Vts[(u * 2 + 1) * 66 + r] = w.h[1];
        }
        if (kb + 64 < SEQ)
            __builtin_prefetch(&K[(b * SEQ + kb + 64 + (tid >> 1)) * D_MODEL + h * HDIM + (tid & 1) * 32], 0, 3);
        __syncthreads();

        // scores = Q . K^T  (four 16-key subtiles x two d k-steps = 8 WMMA)
        v8f s[4] = {};
        #pragma unroll
        for (int kk = 0; kk < 2; kk++) {
            #pragma unroll
            for (int ns = 0; ns < 4; ns++) {
                FragH bf;
                #pragma unroll
                for (int v = 0; v < 8; v++)
                    bf.u[v] = *(const uint32_t*)&Ks[(ns * 16 + lm) * 66 + kk * 32 + kmapB(v, hh)];
                s[ns] = WMMA_F16(qa[kk].v, bf.v, s[ns]);
            }
        }

        // Streaming softmax update; row r = i + 8*hh lives in the 16-lane half
        #pragma unroll
        for (int i = 0; i < 8; i++) {
            float v0 = s[0][i] * 0.125f, v1 = s[1][i] * 0.125f;   // 1/sqrt(64)
            float v2 = s[2][i] * 0.125f, v3 = s[3][i] * 0.125f;
            float mx = fmaxf(fmaxf(v0, v1), fmaxf(v2, v3));
            mx = fmaxf(mx, __shfl_xor(mx, 1, 32));
            mx = fmaxf(mx, __shfl_xor(mx, 2, 32));
            mx = fmaxf(mx, __shfl_xor(mx, 4, 32));
            mx = fmaxf(mx, __shfl_xor(mx, 8, 32));
            float mnew = fmaxf(mrow[i], mx);
            float corr = __expf(mrow[i] - mnew);
            float p0 = __expf(v0 - mnew), p1 = __expf(v1 - mnew);
            float p2 = __expf(v2 - mnew), p3 = __expf(v3 - mnew);
            float rs = (p0 + p1) + (p2 + p3);
            rs += __shfl_xor(rs, 1, 32);
            rs += __shfl_xor(rs, 2, 32);
            rs += __shfl_xor(rs, 4, 32);
            rs += __shfl_xor(rs, 8, 32);
            lrow[i] = lrow[i] * corr + rs;
            mrow[i] = mnew;
            acc[0][i] *= corr; acc[1][i] *= corr; acc[2][i] *= corr; acc[3][i] *= corr;
            // Spill P (C-layout) to per-wave LDS scratch for A-fragment regather
            int prow = (wid * 16 + i + 8 * hh) * 66;
            Ps[prow +      lm] = (_Float16)p0;
            Ps[prow + 16 + lm] = (_Float16)p1;
            Ps[prow + 32 + lm] = (_Float16)p2;
            Ps[prow + 48 + lm] = (_Float16)p3;
        }
        __syncthreads();

        // ctx += P . V   (2 key k-steps x 4 d-subtiles = 8 WMMA, pa reused)
        #pragma unroll
        for (int kk = 0; kk < 2; kk++) {
            FragH pa;
            #pragma unroll
            for (int v = 0; v < 8; v++)
                pa.u[v] = *(const uint32_t*)&Ps[(wid * 16 + lm) * 66 + kk * 32 + kmapA(v, hh)];
            #pragma unroll
            for (int ds = 0; ds < 4; ds++) {
                FragH vb;
                #pragma unroll
                for (int v = 0; v < 8; v++)
                    vb.u[v] = *(const uint32_t*)&Vts[(ds * 16 + lm) * 66 + kk * 32 + kmapB(v, hh)];
                acc[ds] = WMMA_F16(pa.v, vb.v, acc[ds]);
            }
        }
    }

    // Normalize and store ctx (f16, [B,S,H*d] layout)
    #pragma unroll
    for (int i = 0; i < 8; i++) {
        float inv = 1.0f / lrow[i];
        int row  = q0 + wid * 16 + i + 8 * hh;
        int base = (b * SEQ + row) * D_MODEL + h * HDIM;
        #pragma unroll
        for (int ds = 0; ds < 4; ds++)
            C[base + ds * 16 + lm] = (_Float16)(acc[ds][i] * inv);
    }
}

// ---------------------------------------------------------------------------
// Kernel 3: output projection.  ctx[8192x768] (f16) @ wo[768x768] (f32) + bo
// -> f32 output.  Same 128x64 / 32x32-per-wave tiling as kernel 1.
// ---------------------------------------------------------------------------
__global__ __launch_bounds__(256)
void out_proj_kernel(const _Float16* __restrict__ Ctx, const float* __restrict__ W,
                     const float* __restrict__ Bb, float* __restrict__ Out)
{
    __shared__ _Float16 As[128 * 34] __attribute__((aligned(16)));
    __shared__ _Float16 Bs[ 64 * 34] __attribute__((aligned(16)));

    const int m0  = blockIdx.y * 128;
    const int n0  = blockIdx.x * 64;
    const int tid = threadIdx.x;
    const int wid = tid >> 5, lane = tid & 31, lm = lane & 15, hh = lane >> 4;
    const int wm  = wid & 3,  wn   = wid >> 2;

    v8f c00 = {}, c01 = {}, c10 = {}, c11 = {};

    for (int k0 = 0; k0 < D_MODEL; k0 += 32) {
        #pragma unroll
        for (int i = 0; i < 8; i++) {            // 128 rows x 16 half2 = 2048 uints
            int t = tid + i * 256;
            int r = t >> 4, u = t & 15;
            *(uint32_t*)&As[r * 34 + u * 2] =
                *(const uint32_t*)&Ctx[(m0 + r) * D_MODEL + k0 + u * 2];
        }
        #pragma unroll
        for (int i = 0; i < 8; i++) {
            int t = tid + i * 256;
            int kk = t >> 6, nn = t & 63;
            Bs[nn * 34 + kk] = (_Float16)W[(k0 + kk) * D_MODEL + n0 + nn];
        }
        if (k0 + 32 < D_MODEL) {
            __builtin_prefetch(&Ctx[(m0 + (tid >> 1)) * D_MODEL + k0 + 32 + (tid & 1) * 16], 0, 3);
            __builtin_prefetch(&W[(k0 + 32 + (tid >> 3)) * D_MODEL + n0 + (tid & 7) * 8], 0, 3);
        }
        __syncthreads();

        FragH a0, a1, b0, b1;
        #pragma unroll
        for (int v = 0; v < 8; v++) {
            int ka = kmapA(v, hh), kb = kmapB(v, hh);
            a0.u[v] = *(const uint32_t*)&As[(wm * 32 +      lm) * 34 + ka];
            a1.u[v] = *(const uint32_t*)&As[(wm * 32 + 16 + lm) * 34 + ka];
            b0.u[v] = *(const uint32_t*)&Bs[(wn * 32 +      lm) * 34 + kb];
            b1.u[v] = *(const uint32_t*)&Bs[(wn * 32 + 16 + lm) * 34 + kb];
        }
        c00 = WMMA_F16(a0.v, b0.v, c00);
        c01 = WMMA_F16(a0.v, b1.v, c01);
        c10 = WMMA_F16(a1.v, b0.v, c10);
        c11 = WMMA_F16(a1.v, b1.v, c11);
        __syncthreads();
    }

    const int   col0  = n0 + wn * 32 + lm;
    const int   col1  = col0 + 16;
    const float bias0 = Bb[col0];
    const float bias1 = Bb[col1];
    #pragma unroll
    for (int i = 0; i < 8; i++) {
        int row0 = m0 + wm * 32 + i + 8 * hh;
        int row1 = row0 + 16;
        Out[row0 * D_MODEL + col0] = c00[i] + bias0;
        Out[row0 * D_MODEL + col1] = c01[i] + bias1;
        Out[row1 * D_MODEL + col0] = c10[i] + bias0;
        Out[row1 * D_MODEL + col1] = c11[i] + bias1;
    }
}

// ---------------------------------------------------------------------------
extern "C" void kernel_launch(void* const* d_in, const int* in_sizes, int n_in,
                              void* d_out, int out_size, void* d_ws, size_t ws_size,
                              hipStream_t stream)
{
    const float* query = (const float*)d_in[0];
    const float* key   = (const float*)d_in[1];
    const float* value = (const float*)d_in[2];
    const float* wq    = (const float*)d_in[3];
    const float* bq    = (const float*)d_in[4];
    const float* wk    = (const float*)d_in[5];
    const float* bk    = (const float*)d_in[6];
    const float* wv    = (const float*)d_in[7];
    const float* bv    = (const float*)d_in[8];
    const float* wo    = (const float*)d_in[9];
    const float* bo    = (const float*)d_in[10];

    const size_t elems = (size_t)MTOT * D_MODEL;          // 6,291,456
    const size_t bytes = elems * sizeof(uint16_t);        // 12.58 MB per buffer
    if (ws_size < 4 * bytes) return;                      // Qb, Kb, Vb, Ctx

    char* ws = (char*)d_ws;
    _Float16* Qb  = (_Float16*)(ws + 0 * bytes);
    _Float16* Kb  = (_Float16*)(ws + 1 * bytes);
    _Float16* Vb  = (_Float16*)(ws + 2 * bytes);
    _Float16* Ctx = (_Float16*)(ws + 3 * bytes);

    dim3 gProj(D_MODEL / 64, MTOT / 128, 3);              // 12 x 64 x 3
    qkv_proj_kernel<<<gProj, 256, 0, stream>>>(query, key, value,
                                               wq, wk, wv, bq, bk, bv,
                                               Qb, Kb, Vb);

    dim3 gAttn(SEQ / 64, BATCH * NHEAD);                  // 32 x 48
    flash_attn_kernel<<<gAttn, 128, 0, stream>>>(Qb, Kb, Vb, Ctx);

    dim3 gOut(D_MODEL / 64, MTOT / 128);                  // 12 x 64
    out_proj_kernel<<<gOut, 256, 0, stream>>>(Ctx, wo, bo, (float*)d_out);
}
